// R_GCN_48120813584413
// MI455X (gfx1250) — matrix operand order repaired and checked by reading, not compiled, not run on hardware
//
#include <hip/hip_runtime.h>
#include <hip/hip_bf16.h>
#include <stdint.h>

// ---------------------------------------------------------------------------
// R-GCN style graph conv for MI455X (gfx1250, wave32, WMMA).
// Heavy math: v_wmma_f32_16x16x32_bf16 (bf16 in, f32 acc).
// Double-buffered LDS staging via global_load_async_to_lds_b128 (ASYNCcnt),
// so HBM streaming of adjacency tiles overlaps WMMA on the previous tile.
// ---------------------------------------------------------------------------

#ifndef __has_builtin
#define __has_builtin(x) 0
#endif

#if __has_builtin(__builtin_amdgcn_global_load_async_to_lds_b128)
#define ASYNC_LDS 1
#else
#define ASYNC_LDS 0
#endif

typedef __attribute__((ext_vector_type(16))) __bf16 v16bf;
typedef __attribute__((ext_vector_type(8)))  __bf16 v8bf;
typedef __attribute__((ext_vector_type(8)))  float  v8f;
typedef __attribute__((ext_vector_type(4)))  int    v4i;

#define M_NUMS 4096
#define PROJ   256

// 16-byte global -> LDS copy (per lane).
__device__ __forceinline__ void cp16(void* lds, const void* g) {
#if ASYNC_LDS
  __builtin_amdgcn_global_load_async_to_lds_b128(
      (__attribute__((address_space(1))) v4i*)g,
      (__attribute__((address_space(3))) v4i*)lds, 0, 0);
#else
  *(uint4*)lds = *(const uint4*)g;
#endif
}

// Wait for all outstanding async LDS loads, then workgroup barrier.
__device__ __forceinline__ void wait_stage() {
#if ASYNC_LDS
#if __has_builtin(__builtin_amdgcn_s_wait_asynccnt)
  __builtin_amdgcn_s_wait_asynccnt(0);
#else
  asm volatile("s_wait_asynccnt 0" ::: "memory");
#endif
#endif
  __syncthreads();
}

__device__ __forceinline__ v8f wmma_bf16(v16bf a, v16bf b, v8f c) {
  return __builtin_amdgcn_wmma_f32_16x16x32_bf16(false, a, false, b, (short)0, c,
                                                 false, false);
}

// ---------------------------------------------------------------------------
// Prep kernels
// ---------------------------------------------------------------------------
__global__ __launch_bounds__(256) void k_prep_feat(const float* __restrict__ x,
                                                   __bf16* __restrict__ xb) {
  int i = blockIdx.x * 256 + threadIdx.x;
  xb[i] = (__bf16)x[i];
}

// BwT[t][k][n] = W_t[n][k]  (t: 0-2 Wm, 3-5 Wd, 6-9 pair sums of Wx)
__global__ __launch_bounds__(256) void k_prep_w(const float* __restrict__ Wm,
                                                const float* __restrict__ Wd,
                                                const float* __restrict__ Wx,
                                                const float* __restrict__ bxv,
                                                __bf16* __restrict__ BwT,
                                                float* __restrict__ bsum) {
  const int t = blockIdx.y, k = blockIdx.x, n = threadIdx.x;
  float v;
  if (t < 3) {
    v = Wm[t * 65536 + n * 256 + k];
  } else if (t < 6) {
    v = Wd[(t - 3) * 65536 + n * 256 + k];
  } else {
    const int pa[4] = {0, 2, 4, 6};
    const int pb[4] = {1, 5, 3, 7};
    int a = pa[t - 6], b = pb[t - 6];
    v = Wx[a * 65536 + n * 256 + k] + Wx[b * 65536 + n * 256 + k];
    if (k == 0) bsum[(t - 6) * 256 + n] = bxv[a * 256 + n] + bxv[b * 256 + n];
  }
  BwT[t * 65536 + k * 256 + n] = (__bf16)v;
}

// ---------------------------------------------------------------------------
// Transform GEMM: Y = Xb(4096x256 bf16) @ BwT(256x256 bf16) + bias.
// Block = 32M x 256N, 8 waves as 2M x 4N, wave tile 16M x 64N.
// PAIR instantiation keeps two accumulator sets, row-selects by central flag.
// ---------------------------------------------------------------------------
template <bool PAIR>
__global__ __launch_bounds__(256) void k_transform(
    const __bf16* __restrict__ Xb, const __bf16* __restrict__ BwT,
    const float* __restrict__ bm, const float* __restrict__ bd,
    const float* __restrict__ bsum, const int* __restrict__ central_m,
    const int* __restrict__ central_d, float* __restrict__ out,
    __bf16* __restrict__ Tm1b, __bf16* __restrict__ Tm2b,
    __bf16* __restrict__ Td1b, __bf16* __restrict__ Td2b,
    __bf16* __restrict__ SmA, __bf16* __restrict__ SmB,
    __bf16* __restrict__ SdA, __bf16* __restrict__ SdB) {
  const int op = PAIR ? (blockIdx.y + 6) : blockIdx.y;
  const __bf16* X = Xb;
  const __bf16* B0 = nullptr;
  const __bf16* B1 = nullptr;
  const float* bias0 = nullptr;
  const float* bias1 = nullptr;
  const int* cen = nullptr;
  float* fout = nullptr;
  __bf16* hout = nullptr;
  const __bf16* Xd = Xb + (size_t)M_NUMS * PROJ;
  switch (op) {
    case 0: X = Xb; B0 = BwT;             bias0 = bm;        fout = out;                       break;
    case 1: X = Xb; B0 = BwT + 1 * 65536; bias0 = bm + 256;  hout = Tm1b;                      break;
    case 2: X = Xb; B0 = BwT + 2 * 65536; bias0 = bm + 512;  hout = Tm2b;                      break;
    case 3: X = Xd; B0 = BwT + 3 * 65536; bias0 = bd;        fout = out + (size_t)M_NUMS * PROJ; break;
    case 4: X = Xd; B0 = BwT + 4 * 65536; bias0 = bd + 256;  hout = Td1b;                      break;
    case 5: X = Xd; B0 = BwT + 5 * 65536; bias0 = bd + 512;  hout = Td2b;                      break;
    case 6: X = Xd; B0 = BwT + 6 * 65536; B1 = BwT + 7 * 65536; bias0 = bsum;       bias1 = bsum + 256; cen = central_d; hout = SmA; break;
    case 7: X = Xd; B0 = BwT + 8 * 65536; B1 = BwT + 9 * 65536; bias0 = bsum + 512; bias1 = bsum + 768; cen = central_d; hout = SmB; break;
    case 8: X = Xb; B0 = BwT + 6 * 65536; B1 = BwT + 8 * 65536; bias0 = bsum;       bias1 = bsum + 512; cen = central_m; hout = SdA; break;
    default:X = Xb; B0 = BwT + 7 * 65536; B1 = BwT + 9 * 65536; bias0 = bsum + 256; bias1 = bsum + 768; cen = central_m; hout = SdB; break;
  }

  const int i0 = blockIdx.x * 32;
  const int tid = threadIdx.x;
  const int lane = tid & 31, w = tid >> 5, wm = w >> 2, wn = w & 3;
  const int r = lane & 15, h = lane >> 4;

  __shared__ __align__(64) __bf16 sX[2][32 * 32];
  __shared__ __align__(64) __bf16 sB0[2][32 * 256];
  __shared__ __align__(64) __bf16 sB1[PAIR ? 2 : 1][PAIR ? 32 * 256 : 1];

  v8f acc0[4] = {};
  v8f acc1[4] = {};

  auto stage = [&](int it, int bsel) {
    const int k0 = it * 32;
    if (tid < 128) {
      int row = tid >> 2, c8 = (tid & 3) * 8;
      cp16(&sX[bsel][row * 32 + c8], X + (size_t)(i0 + row) * 256 + k0 + c8);
    }
#pragma unroll
    for (int q = 0; q < 4; ++q) {
      int flat = q * 256 + tid;
      int row = flat >> 5, c8 = (flat & 31) * 8;
      cp16(&sB0[bsel][row * 256 + c8], B0 + (size_t)(k0 + row) * 256 + c8);
      if (PAIR)
        cp16(&sB1[bsel][row * 256 + c8], B1 + (size_t)(k0 + row) * 256 + c8);
    }
  };

  const int niter = 256 / 32;
  stage(0, 0);
  wait_stage();

  for (int it = 0; it < niter; ++it) {
    const int bsel = it & 1;
    if (it + 1 < niter) stage(it + 1, bsel ^ 1);

    // A fragment: lane holds row (wm*16 + r), K = {8h..8h+7, 16+8h..16+8h+7}
    int rb = wm * 16 + r;
    v16bf a;
    const __bf16* ap = &sX[bsel][rb * 32 + 8 * h];
    *(v8bf*)&a = *(const v8bf*)ap;
    *(((v8bf*)&a) + 1) = *(const v8bf*)(ap + 16);

#pragma unroll
    for (int u = 0; u < 4; ++u) {
      v16bf b0;
      for (int s = 0; s < 16; ++s) b0[s] = sB0[bsel][((s & 8) * 2 + 8 * h + (s & 7)) * 256 + wn * 64 + u * 16 + r];
      acc0[u] = wmma_bf16(a, b0, acc0[u]);
      if (PAIR) {
        v16bf b1;
        for (int s = 0; s < 16; ++s) b1[s] = sB1[bsel][((s & 8) * 2 + 8 * h + (s & 7)) * 256 + wn * 64 + u * 16 + r];
        acc1[u] = wmma_bf16(a, b1, acc1[u]);
      }
    }
    wait_stage();
  }

#pragma unroll
  for (int u = 0; u < 4; ++u) {
    int col = wn * 64 + u * 16 + r;
#pragma unroll
    for (int v = 0; v < 8; ++v) {
      int row = i0 + wm * 16 + v + 8 * h;
      float val = acc0[u][v] + bias0[col];
      if (PAIR) {
        float val1 = acc1[u][v] + bias1[col];
        val = (cen[row] != 0) ? val : val1;
      }
      if (fout) fout[(size_t)row * 256 + col] = val;
      else      hout[(size_t)row * 256 + col] = (__bf16)val;
    }
  }
}

// ---------------------------------------------------------------------------
// Homog: out += A0 @ T1 + A1 @ T2, A generated on the fly from co + rsqrt(deg).
// Block = 64M x 256N, 8 waves as 2M x 4N, wave tile 32M x 64N (one acc set,
// two WMMAs per k-step feed the same accumulator).  Double-buffered.
// ---------------------------------------------------------------------------
__global__ __launch_bounds__(256) void k_homog(
    const float* __restrict__ m_co, const float* __restrict__ d_co,
    const float* __restrict__ m_deg, const float* __restrict__ d_deg,
    const __bf16* __restrict__ Tm1, const __bf16* __restrict__ Tm2,
    const __bf16* __restrict__ Td1, const __bf16* __restrict__ Td2,
    float* __restrict__ out) {
  const int side = blockIdx.y;
  const float* co = side ? d_co : m_co;
  const float* deg = side ? d_deg : m_deg;
  const __bf16* B1g = side ? Td1 : Tm1;
  const __bf16* B2g = side ? Td2 : Tm2;
  float* o = out + (size_t)side * M_NUMS * PROJ;

  const int i0 = blockIdx.x * 64;
  const int tid = threadIdx.x;
  const int lane = tid & 31, w = tid >> 5, wm = w >> 2, wn = w & 3;
  const int r = lane & 15, h = lane >> 4;

  __shared__ __align__(64) float  sCo[2][64 * 32];
  __shared__ __align__(64) __bf16 sB1[2][32 * 256];
  __shared__ __align__(64) __bf16 sB2[2][32 * 256];
  __shared__ float sR0[64], sR1[64];
  __shared__ float sC0[2][32], sC1[2][32];

  if (tid < 64) {
    sR0[tid] = rsqrtf(deg[(i0 + tid) * 2 + 0]);
    sR1[tid] = rsqrtf(deg[(i0 + tid) * 2 + 1]);
  }

  auto stage = [&](int it, int bsel) {
    const int k0 = it * 32;
    if (tid < 32) {
      sC0[bsel][tid] = rsqrtf(deg[(k0 + tid) * 2 + 0]);
      sC1[bsel][tid] = rsqrtf(deg[(k0 + tid) * 2 + 1]);
    }
#pragma unroll
    for (int q = 0; q < 2; ++q) {
      int flat = q * 256 + tid;
      int row = flat >> 3, c4 = (flat & 7) * 4;
      cp16(&sCo[bsel][row * 32 + c4], co + (size_t)(i0 + row) * 4096 + k0 + c4);
    }
#pragma unroll
    for (int q = 0; q < 4; ++q) {
      int flat = q * 256 + tid;
      int row = flat >> 5, c8 = (flat & 31) * 8;
      cp16(&sB1[bsel][row * 256 + c8], B1g + (size_t)(k0 + row) * 256 + c8);
      cp16(&sB2[bsel][row * 256 + c8], B2g + (size_t)(k0 + row) * 256 + c8);
    }
  };

  stage(0, 0);
  wait_stage();

  float rs0[2], rs1[2];
#pragma unroll
  for (int t = 0; t < 2; ++t) {
    int rb = wm * 32 + t * 16 + r;
    rs0[t] = sR0[rb];
    rs1[t] = sR1[rb];
  }

  v8f acc[2][4] = {};

  const int niter = 4096 / 32;
  for (int it = 0; it < niter; ++it) {
    const int bsel = it & 1;
    const int k0 = it * 32;
    if (it + 1 < niter) stage(it + 1, bsel ^ 1);

#pragma unroll
    for (int t = 0; t < 2; ++t) {
      int rb = wm * 32 + t * 16 + r;
      int ig = i0 + rb;
      v16bf a0, a1;
#pragma unroll
      for (int half = 0; half < 2; ++half) {
        int kb = half * 16 + 8 * h;
#pragma unroll
        for (int j = 0; j < 8; ++j) {
          int k = kb + j;
          float c = sCo[bsel][rb * 32 + k];
          bool offd = (ig != (k0 + k));
          float e0 = (offd && (c != 0.f)) ? rs0[t] * sC0[bsel][k] : 0.f;
          float e1 = (offd && (c == 0.f)) ? rs1[t] * sC1[bsel][k] : 0.f;
          a0[half * 8 + j] = (__bf16)e0;
          a1[half * 8 + j] = (__bf16)e1;
        }
      }
#pragma unroll
      for (int u = 0; u < 4; ++u) {
        v16bf b1, b2;
        for (int s = 0; s < 16; ++s) {
          b1[s] = sB1[bsel][((s & 8) * 2 + 8 * h + (s & 7)) * 256 + wn * 64 + u * 16 + r];
          b2[s] = sB2[bsel][((s & 8) * 2 + 8 * h + (s & 7)) * 256 + wn * 64 + u * 16 + r];
        }
        acc[t][u] = wmma_bf16(a0, b1, acc[t][u]);
        acc[t][u] = wmma_bf16(a1, b2, acc[t][u]);
      }
    }
    wait_stage();
  }

#pragma unroll
  for (int t = 0; t < 2; ++t)
#pragma unroll
    for (int u = 0; u < 4; ++u) {
      int col = wn * 64 + u * 16 + r;
#pragma unroll
      for (int v = 0; v < 8; ++v) {
        int row = i0 + wm * 32 + t * 16 + v + 8 * h;
        o[(size_t)row * 256 + col] += acc[t][u][v];
      }
    }
}

// ---------------------------------------------------------------------------
// Cross: out += 0.5/deg_row * (assoc @ (central_row ? SA : SB)).
// Block = 32M x 256N, waves 2M x 4N, wave tile 16M x 64N, two acc sets.
// Double-buffered.
// ---------------------------------------------------------------------------
__global__ __launch_bounds__(256) void k_cross(
    const float* __restrict__ bigraph, const __bf16* __restrict__ SmA,
    const __bf16* __restrict__ SmB, const __bf16* __restrict__ SdA,
    const __bf16* __restrict__ SdB, const int* __restrict__ central_m,
    const int* __restrict__ central_d, const float* __restrict__ m_d_degree,
    const float* __restrict__ d_m_degree, float* __restrict__ out) {
  const int side = blockIdx.y;
  const __bf16* BA = side ? SdA : SmA;
  const __bf16* BB = side ? SdB : SmB;
  const int* cen = side ? central_d : central_m;
  const float* deg = side ? d_m_degree : m_d_degree;
  float* o = out + (size_t)side * M_NUMS * PROJ;
  const float* asrc = side ? (bigraph + (size_t)M_NUMS * 8192) : (bigraph + M_NUMS);

  const int i0 = blockIdx.x * 32;
  const int tid = threadIdx.x;
  const int lane = tid & 31, w = tid >> 5, wm = w >> 2, wn = w & 3;
  const int r = lane & 15, h = lane >> 4;

  __shared__ __align__(64) float  sA[2][32 * 32];
  __shared__ __align__(64) __bf16 sBA[2][32 * 256];
  __shared__ __align__(64) __bf16 sBB[2][32 * 256];

  v8f accA[4] = {};
  v8f accB[4] = {};

  auto stage = [&](int it, int bsel) {
    const int k0 = it * 32;
    {
      int row = tid >> 3, c4 = (tid & 7) * 4;
      cp16(&sA[bsel][row * 32 + c4], asrc + (size_t)(i0 + row) * 8192 + k0 + c4);
    }
#pragma unroll
    for (int q = 0; q < 4; ++q) {
      int flat = q * 256 + tid;
      int row = flat >> 5, c8 = (flat & 31) * 8;
      cp16(&sBA[bsel][row * 256 + c8], BA + (size_t)(k0 + row) * 256 + c8);
      cp16(&sBB[bsel][row * 256 + c8], BB + (size_t)(k0 + row) * 256 + c8);
    }
  };

  stage(0, 0);
  wait_stage();

  const int niter = 4096 / 32;
  for (int it = 0; it < niter; ++it) {
    const int bsel = it & 1;
    if (it + 1 < niter) stage(it + 1, bsel ^ 1);

    int rb = wm * 16 + r;
    v16bf a;
#pragma unroll
    for (int half = 0; half < 2; ++half) {
      int kb = half * 16 + 8 * h;
#pragma unroll
      for (int j = 0; j < 8; ++j)
        a[half * 8 + j] = (__bf16)sA[bsel][rb * 32 + kb + j];
    }
#pragma unroll
    for (int u = 0; u < 4; ++u) {
      v16bf ba, bb;
      for (int s = 0; s < 16; ++s) {
        ba[s] = sBA[bsel][((s & 8) * 2 + 8 * h + (s & 7)) * 256 + wn * 64 + u * 16 + r];
        bb[s] = sBB[bsel][((s & 8) * 2 + 8 * h + (s & 7)) * 256 + wn * 64 + u * 16 + r];
      }
      accA[u] = wmma_bf16(a, ba, accA[u]);
      accB[u] = wmma_bf16(a, bb, accB[u]);
    }
    wait_stage();
  }

#pragma unroll
  for (int u = 0; u < 4; ++u) {
    int col = wn * 64 + u * 16 + r;
#pragma unroll
    for (int v = 0; v < 8; ++v) {
      int row = i0 + wm * 16 + v + 8 * h;
      float val = (cen[row] != 0) ? accA[u][v] : accB[u][v];
      o[(size_t)row * 256 + col] += val * (0.5f / deg[row]);
    }
  }
}

// ---------------------------------------------------------------------------
// Launch
// ---------------------------------------------------------------------------
extern "C" void kernel_launch(void* const* d_in, const int* in_sizes, int n_in,
                              void* d_out, int out_size, void* d_ws,
                              size_t ws_size, hipStream_t stream) {
  const float* m_d_feat   = (const float*)d_in[0];
  const float* bigraph    = (const float*)d_in[1];
  const float* m_co       = (const float*)d_in[2];
  const float* d_co       = (const float*)d_in[3];
  const float* m_co_deg   = (const float*)d_in[4];
  const float* d_co_deg   = (const float*)d_in[5];
  const float* m_d_degree = (const float*)d_in[6];
  const float* d_m_degree = (const float*)d_in[7];
  const float* Wm = (const float*)d_in[8];
  const float* bm = (const float*)d_in[9];
  const float* Wd = (const float*)d_in[10];
  const float* bd = (const float*)d_in[11];
  const float* Wx = (const float*)d_in[12];
  const float* bx = (const float*)d_in[13];
  const int* central_m = (const int*)d_in[14];
  const int* central_d = (const int*)d_in[15];
  float* out = (float*)d_out;

  // workspace carve-up (~22.3 MB total)
  char* wsp = (char*)d_ws;
  auto carve = [&](size_t bytes) {
    void* p = (void*)wsp;
    wsp += (bytes + 255) & ~(size_t)255;
    return p;
  };
  const size_t NB = (size_t)M_NUMS * PROJ * sizeof(__bf16);
  __bf16* Xb   = (__bf16*)carve((size_t)8192 * PROJ * sizeof(__bf16));
  __bf16* BwT  = (__bf16*)carve((size_t)10 * 256 * 256 * sizeof(__bf16));
  float*  bsum = (float*)carve((size_t)4 * 256 * sizeof(float));
  __bf16* Tm1b = (__bf16*)carve(NB);
  __bf16* Tm2b = (__bf16*)carve(NB);
  __bf16* Td1b = (__bf16*)carve(NB);
  __bf16* Td2b = (__bf16*)carve(NB);
  __bf16* SmA  = (__bf16*)carve(NB);
  __bf16* SmB  = (__bf16*)carve(NB);
  __bf16* SdA  = (__bf16*)carve(NB);
  __bf16* SdB  = (__bf16*)carve(NB);

  k_prep_feat<<<8192, 256, 0, stream>>>(m_d_feat, Xb);
  k_prep_w<<<dim3(256, 10), 256, 0, stream>>>(Wm, Wd, Wx, bx, BwT, bsum);
  k_transform<false><<<dim3(128, 6), 256, 0, stream>>>(
      Xb, BwT, bm, bd, bsum, central_m, central_d, out, Tm1b, Tm2b, Td1b, Td2b,
      SmA, SmB, SdA, SdB);
  k_transform<true><<<dim3(128, 4), 256, 0, stream>>>(
      Xb, BwT, bm, bd, bsum, central_m, central_d, out, Tm1b, Tm2b, Td1b, Td2b,
      SmA, SmB, SdA, SdB);
  k_homog<<<dim3(64, 2), 256, 0, stream>>>(m_co, d_co, m_co_deg, d_co_deg,
                                           Tm1b, Tm2b, Td1b, Td2b, out);
  k_cross<<<dim3(128, 2), 256, 0, stream>>>(bigraph, SmA, SmB, SdA, SdB,
                                            central_m, central_d, m_d_degree,
                                            d_m_degree, out);
}
